// GCN_22797686407974
// MI455X (gfx1250) — matrix operand order repaired and checked by reading, not compiled, not run on hardware
//
#include <hip/hip_runtime.h>
#include <stdint.h>

// Problem constants (match reference)
#define NN    8192   // nodes
#define FEAT  128    // input features
#define HID   256    // hidden
#define NCLS  64     // classes
#define BLK   256    // nodes per env block (adj is block-diagonal)
#define DEG   16     // max degree

typedef __bf16 bf16_t;
typedef __attribute__((ext_vector_type(16))) bf16_t   v16bf;
typedef __attribute__((ext_vector_type(8)))  float    v8f;
typedef __attribute__((ext_vector_type(4)))  uint32_t u32x4;
typedef __attribute__((ext_vector_type(8)))  int32_t  i32x8;
typedef __attribute__((ext_vector_type(4)))  int32_t  i32x4;

#if defined(__has_builtin)
#if __has_builtin(__builtin_amdgcn_tensor_load_to_lds) && __has_builtin(__builtin_amdgcn_s_wait_tensorcnt)
#define HAVE_TDM 1
#endif
#endif
#ifndef HAVE_TDM
#define HAVE_TDM 0
#endif

// ---- bf16 <-> f32 helpers (storage as uint16_t; RNE rounding) ----
__device__ inline uint16_t f2bf(float f) {
  uint32_t u = __builtin_bit_cast(uint32_t, f);
  u += 0x7FFFu + ((u >> 16) & 1u);
  return (uint16_t)(u >> 16);
}
__device__ inline float bf2f(uint16_t v) {
  return __builtin_bit_cast(float, (uint32_t)v << 16);
}

// Build a 16-element bf16 WMMA fragment from two 128-bit chunks.
union FragCvt { uint4 u4[2]; v16bf v; };
__device__ inline v16bf make_frag(uint4 lo, uint4 hi) {
  FragCvt f; f.u4[0] = lo; f.u4[1] = hi; return f.v;
}

// ---- Tensor Data Mover: 2D tile (rows x elems_per_row) global -> LDS ----
// data_size_code: 0=1B, 1=2B, 2=4B, 3=8B
__device__ inline void tdm_load_tile_2d(uint32_t lds_byte_off, const void* gptr,
                                        uint32_t elems_per_row, uint32_t rows,
                                        uint32_t row_stride_elems, uint32_t data_size_code) {
#if HAVE_TDM
  uint64_t ga = (uint64_t)(uintptr_t)gptr;
  u32x4 g0;
  g0[0] = 1u;                                    // count=1 (valid descriptor)
  g0[1] = lds_byte_off;                          // lds_addr (bytes)
  g0[2] = (uint32_t)ga;                          // global_addr[31:0]
  g0[3] = ((uint32_t)(ga >> 32) & 0x01FFFFFFu)   // global_addr[56:32]
          | (2u << 30);                          // type=2 ("image")
  i32x8 g1;
  g1[0] = (int32_t)(data_size_code << 16);       // wg_mask=0, data_size
  g1[1] = (int32_t)((elems_per_row & 0xFFFFu) << 16);               // tensor_dim0[15:0]
  g1[2] = (int32_t)(((elems_per_row >> 16) & 0xFFFFu)
                    | ((rows & 0xFFFFu) << 16));                    // dim0[31:16] | dim1[15:0]
  g1[3] = (int32_t)(((rows >> 16) & 0xFFFFu)
                    | ((elems_per_row & 0xFFFFu) << 16));           // dim1[31:16] | tile_dim0
  g1[4] = (int32_t)(rows & 0xFFFFu);             // tile_dim1, tile_dim2=0
  g1[5] = (int32_t)row_stride_elems;             // tensor_dim0_stride[31:0]
  g1[6] = 0;                                     // stride0[47:32] | stride1 lo
  g1[7] = 0;
  i32x4 gz4 = {0, 0, 0, 0};
  i32x8 gz8 = {0, 0, 0, 0, 0, 0, 0, 0};
  __builtin_amdgcn_tensor_load_to_lds(g0, g1, gz4, gz4, gz8, 0);
#else
  (void)lds_byte_off; (void)gptr; (void)elems_per_row; (void)rows;
  (void)row_stride_elems; (void)data_size_code;
#endif
}

__device__ inline void tdm_wait_all() {
#if HAVE_TDM
  __builtin_amdgcn_s_wait_tensorcnt(0);
#endif
}

// ---- K0: f32 -> bf16 conversion ----
__global__ void f32_to_bf16_k(const float* __restrict__ in, uint16_t* __restrict__ out, int n) {
  int i = blockIdx.x * blockDim.x + threadIdx.x;
  if (i < n) out[i] = f2bf(in[i]);
}

// ---- K1/K3: bf16 WMMA GEMM. C[M,N] = A[M,K] * B[K,N], one wave per 16x16 tile ----
__global__ __launch_bounds__(32) void gemm_bf16(const uint16_t* __restrict__ A,
                                                const uint16_t* __restrict__ B,
                                                uint16_t* __restrict__ C,
                                                int M, int K, int N) {
  const int tm   = blockIdx.x * 16;
  const int tn   = blockIdx.y * 16;
  const int lane = threadIdx.x;      // 0..31 (wave32)
  const int hi   = lane >> 4;        // lane group
  const int lr   = lane & 15;

  v8f acc = {};
  for (int k0 = 0; k0 < K; k0 += 32) {
    // A fragment (16x32 bf16): lane -> row M=tm+lr; hi lanes hold K+8.
    // elems 0..7 = K[k0+8*hi .. +8), elems 8..15 = K[k0+16+8*hi .. +8)
    const uint16_t* ap = A + (size_t)(tm + lr) * K + (k0 + 8 * hi);
    uint4 alo = *(const uint4*)(ap);
    uint4 ahi = *(const uint4*)(ap + 16);
    // B fragment (32x16 bf16): lane -> column N=tn+lr; elems in same K order as A.
    const uint16_t* bp = B + (size_t)(k0 + 8 * hi) * N + (tn + lr);
    v16bf bv;
    for (int s = 0; s < 16; ++s)
      bv[s] = __builtin_bit_cast(bf16_t, bp[(size_t)(s < 8 ? s : s + 8) * N]);
    v16bf av = make_frag(alo, ahi);
    acc = __builtin_amdgcn_wmma_f32_16x16x32_bf16(
        /*neg_a=*/false, av, /*neg_b=*/false, bv,
        /*c_mod=*/(short)0, acc, /*reuse_a=*/false, /*reuse_b=*/false);
  }
  // C/D layout: VGPR r -> M = tm + r + 8*hi, N = tn + lr
  for (int r = 0; r < 8; ++r)
    C[(size_t)(tm + 8 * hi + r) * N + (tn + lr)] = f2bf(acc[r]);
}

// ---- K2: per-block dedup'd aggregation + ReLU.  h = relu(adj_blk @ XW1_blk) ----
__global__ __launch_bounds__(256) void aggregate1(const uint16_t* __restrict__ xw1b,
                                                  const int* __restrict__ edges,
                                                  uint16_t* __restrict__ hb) {
  __shared__ uint16_t tile[BLK * HID];   // 128 KB: XW1 block (bf16)
  __shared__ int      eidx[BLK * DEG];   // 16 KB
  __shared__ uint32_t emask[BLK];        // 1 KB: first-occurrence masks
  const int b = blockIdx.x;
  const int t = threadIdx.x;
  const uint16_t* src = xw1b + (size_t)b * BLK * HID;

#if HAVE_TDM
  if (t == 0)
    tdm_load_tile_2d((uint32_t)(uintptr_t)&tile[0], src, HID, BLK, HID, /*2B*/1u);
#else
  {
    const uint4* s4 = (const uint4*)src;
    uint4* d4 = (uint4*)&tile[0];
    for (int i = t; i < BLK * HID * 2 / 16; i += 256) d4[i] = s4[i];
  }
#endif

  // Stage this row's edge list and compute dedup mask (duplicates count once).
  int e[DEG];
  const int* erow = edges + (size_t)(b * BLK + t) * DEG;
  for (int i = 0; i < DEG; ++i) { e[i] = erow[i]; eidx[t * DEG + i] = e[i]; }
  uint32_t m = 0;
  for (int i = 0; i < DEG; ++i) {
    bool dup = (e[i] < 0) || (e[i] >= BLK);
    for (int j = 0; j < i; ++j) dup = dup || (e[j] == e[i]);
    if (!dup) m |= (1u << i);
  }
  emask[t] = m;

#if HAVE_TDM
  if (t == 0) tdm_wait_all();
#endif
  __syncthreads();

  const int c = t;  // each thread owns one feature column (HID == 256 == blockDim)
  for (int r = 0; r < BLK; ++r) {
    const uint32_t mm = emask[r];
    const int* er = &eidx[r * DEG];
    float s = 0.f;
    for (int i = 0; i < DEG; ++i)
      if (mm & (1u << i)) s += bf2f(tile[er[i] * HID + c]);
    hb[((size_t)b * BLK + r) * HID + c] = f2bf(fmaxf(s, 0.f));
  }
}

// ---- K4: per-block dedup'd aggregation + fused log_softmax ----
__global__ __launch_bounds__(256) void aggregate2(const uint16_t* __restrict__ hw2b,
                                                  const int* __restrict__ edges,
                                                  float* __restrict__ out) {
  __shared__ uint16_t tile[BLK * NCLS];  // 32 KB: HW2 block (bf16)
  const int b = blockIdx.x;
  const int t = threadIdx.x;             // row within block
  const uint16_t* src = hw2b + (size_t)b * BLK * NCLS;

#if HAVE_TDM
  if (t == 0)
    tdm_load_tile_2d((uint32_t)(uintptr_t)&tile[0], src, NCLS, BLK, NCLS, /*2B*/1u);
#else
  {
    const uint4* s4 = (const uint4*)src;
    uint4* d4 = (uint4*)&tile[0];
    for (int i = t; i < BLK * NCLS * 2 / 16; i += 256) d4[i] = s4[i];
  }
#endif

  int e[DEG];
  const int* erow = edges + (size_t)(b * BLK + t) * DEG;
  for (int i = 0; i < DEG; ++i) e[i] = erow[i];
  uint32_t m = 0;
  for (int i = 0; i < DEG; ++i) {
    bool dup = (e[i] < 0) || (e[i] >= BLK);
    for (int j = 0; j < i; ++j) dup = dup || (e[j] == e[i]);
    if (!dup) m |= (1u << i);
  }

#if HAVE_TDM
  if (t == 0) tdm_wait_all();
#endif
  __syncthreads();

  float acc[NCLS];
  for (int j = 0; j < NCLS; ++j) acc[j] = 0.f;
  for (int i = 0; i < DEG; ++i) {
    if (m & (1u << i)) {
      const uint16_t* trow = &tile[e[i] * NCLS];
      for (int j = 0; j < NCLS; ++j) acc[j] += bf2f(trow[j]);
    }
  }
  // log_softmax over the 64 classes
  float mx = acc[0];
  for (int j = 1; j < NCLS; ++j) mx = fmaxf(mx, acc[j]);
  float s = 0.f;
  for (int j = 0; j < NCLS; ++j) s += __expf(acc[j] - mx);
  const float lse = __logf(s);
  float* orow = out + (size_t)(b * BLK + t) * NCLS;
  for (int j = 0; j < NCLS; ++j) orow[j] = acc[j] - mx - lse;
}

// ---- host launcher ----
extern "C" void kernel_launch(void* const* d_in, const int* in_sizes, int n_in,
                              void* d_out, int out_size, void* d_ws, size_t ws_size,
                              hipStream_t stream) {
  (void)in_sizes; (void)n_in; (void)out_size; (void)ws_size;
  const float* x    = (const float*)d_in[0];
  const int*   edges = (const int*)d_in[1];
  const float* W1   = (const float*)d_in[2];
  const float* W2   = (const float*)d_in[3];
  float* out = (float*)d_out;

  char* ws = (char*)d_ws;
  // workspace layout (bf16 intermediates), ~11.1 MB total
  uint16_t* xb   = (uint16_t*)(ws);                               // 8192*128*2 = 2 MB
  uint16_t* w1b  = (uint16_t*)(ws + 2097152);                     // 64 KB
  uint16_t* w2b  = (uint16_t*)(ws + 2097152 + 65536);             // 32 KB
  uint16_t* xw1b = (uint16_t*)(ws + 2097152 + 65536 + 32768);     // 4 MB
  uint16_t* hb   = (uint16_t*)(ws + 2097152 + 65536 + 32768 + 4194304);           // 4 MB
  uint16_t* hw2b = (uint16_t*)(ws + 2097152 + 65536 + 32768 + 4194304 + 4194304); // 1 MB

  f32_to_bf16_k<<<(NN * FEAT + 255) / 256, 256, 0, stream>>>(x,  xb,  NN * FEAT);
  f32_to_bf16_k<<<(FEAT * HID + 255) / 256, 256, 0, stream>>>(W1, w1b, FEAT * HID);
  f32_to_bf16_k<<<(HID * NCLS + 255) / 256, 256, 0, stream>>>(W2, w2b, HID * NCLS);

  // XW1 = x @ W1   (8192x128 @ 128x256)
  gemm_bf16<<<dim3(NN / 16, HID / 16), 32, 0, stream>>>(xb, w1b, xw1b, NN, FEAT, HID);
  // h = relu(adj @ XW1), block-diagonal aggregation with dedup
  aggregate1<<<NN / BLK, BLK, 0, stream>>>(xw1b, edges, hb);
  // HW2 = h @ W2   (8192x256 @ 256x64)
  gemm_bf16<<<dim3(NN / 16, NCLS / 16), 32, 0, stream>>>(hb, w2b, hw2b, NN, HID, NCLS);
  // out = log_softmax(adj @ HW2)
  aggregate2<<<NN / BLK, BLK, 0, stream>>>(hw2b, edges, out);
}
